// DeepGEMM_Linear_62569083568919
// MI455X (gfx1250) — 8-bit matrix operand order repaired and checked by reading, compiled, not run on hardware
//
#include <hip/hip_runtime.h>
#include <hip/hip_bf16.h>


typedef __attribute__((ext_vector_type(16))) int   v16i_t;
typedef __attribute__((ext_vector_type(8)))  float v8f_t;
typedef __attribute__((ext_vector_type(4)))  int   v4i_t;
typedef __attribute__((ext_vector_type(4)))  float v4f_t;
typedef __attribute__((ext_vector_type(2)))  float v2f_t;

typedef __attribute__((address_space(1))) v4i_t g_v4i;
typedef __attribute__((address_space(3))) v4i_t l_v4i;

#define BATCH    16384
#define IN_F     4000
#define IN_PAD   4096
#define OUT_F    4000
#define OUT_ROWS 4032
#define N_PAD    4096
#define SBLK     32
#define TILE     128
#define LSTR     144

#if __has_builtin(__builtin_amdgcn_global_load_async_to_lds_b128) && \
    __has_builtin(__builtin_amdgcn_s_wait_asynccnt)
#define USE_ASYNC_COPY 1
#else
#define USE_ASYNC_COPY 0
#endif

__device__ __forceinline__ unsigned short f32_to_bf16_rne(float f) {
  unsigned int u = __float_as_uint(f);
  if ((u & 0x7FFFFFFFu) > 0x7F800000u) return (unsigned short)((u >> 16) | 0x0040u);
  u += 0x7FFFu + ((u >> 16) & 1u);
  return (unsigned short)(u >> 16);
}
__device__ __forceinline__ float bf16_to_f32(unsigned short h) {
  return __uint_as_float(((unsigned int)h) << 16);
}

#if __has_builtin(__builtin_amdgcn_cvt_pk_fp8_f32)
__device__ __forceinline__ unsigned int pack4_e4m3(float f0, float f1, float f2, float f3) {
  int p = 0;
  p = __builtin_amdgcn_cvt_pk_fp8_f32(f0, f1, p, false);
  p = __builtin_amdgcn_cvt_pk_fp8_f32(f2, f3, p, true);
  return (unsigned int)p;
}
#else
__device__ __forceinline__ unsigned char f32_to_e4m3_sw(float f) {
  unsigned int u    = __float_as_uint(f);
  unsigned int s    = (u >> 24) & 0x80u;
  unsigned int absu = u & 0x7FFFFFFFu;
  if (absu > 0x7F800000u) return (unsigned char)(s | 0x7Fu);
  if (absu >= 0x43E80000u) return (unsigned char)(s | 0x7Eu);
  int e = (int)((absu >> 23) & 0xFFu) - 127;
  if (e >= -6) {
    unsigned int man = absu & 0x7FFFFFu;
    unsigned int m3  = man >> 20;
    unsigned int rem = man & 0xFFFFFu;
    m3 += (rem > 0x80000u || (rem == 0x80000u && (m3 & 1u))) ? 1u : 0u;
    unsigned int e4 = (unsigned int)(e + 7);
    if (m3 == 8u) { m3 = 0u; e4 += 1u; }
    if (e4 > 15u || (e4 == 15u && m3 == 7u)) return (unsigned char)(s | 0x7Eu);
    return (unsigned char)(s | (e4 << 3) | m3);
  }
  float af = __uint_as_float(absu);
  int q = (int)rintf(af * 512.0f);
  if (q >= 8) return (unsigned char)(s | 0x08u);
  return (unsigned char)(s | (unsigned int)q);
}
__device__ __forceinline__ unsigned int pack4_e4m3(float f0, float f1, float f2, float f3) {
  return (unsigned int)f32_to_e4m3_sw(f0)         | ((unsigned int)f32_to_e4m3_sw(f1) << 8) |
         ((unsigned int)f32_to_e4m3_sw(f2) << 16) | ((unsigned int)f32_to_e4m3_sw(f3) << 24);
}
#endif

__global__ void quant_x_kernel(const float* __restrict__ x, unsigned char* __restrict__ xq) {
  unsigned int gid = blockIdx.x * blockDim.x + threadIdx.x;
  unsigned int row = gid >> 10;
  unsigned int k4  = (gid & 1023u) << 2;
  const float* xr = x + (size_t)row * IN_F;
  float f0 = 0.f, f1 = 0.f, f2 = 0.f, f3 = 0.f;
  if (k4 + 3 < IN_F) {
    v4f_t v = __builtin_nontemporal_load((const v4f_t*)(xr + k4));
    f0 = v.x; f1 = v.y; f2 = v.z; f3 = v.w;
  } else {
    if (k4 + 0 < IN_F) f0 = xr[k4 + 0];
    if (k4 + 1 < IN_F) f1 = xr[k4 + 1];
    if (k4 + 2 < IN_F) f2 = xr[k4 + 2];
    if (k4 + 3 < IN_F) f3 = xr[k4 + 3];
  }
  *(unsigned int*)(xq + (size_t)row * IN_PAD + k4) = pack4_e4m3(f0, f1, f2, f3);
}

__global__ void quant_w_kernel(const float* __restrict__ w, unsigned char* __restrict__ wq) {
  unsigned int gid = blockIdx.x * blockDim.x + threadIdx.x;
  unsigned int row = gid >> 10;
  unsigned int k4  = (gid & 1023u) << 2;
  unsigned int pk  = 0u;
  if (row < OUT_ROWS) {
    v4f_t v = __builtin_nontemporal_load((const v4f_t*)(w + (size_t)row * IN_PAD + k4));
    pk = pack4_e4m3(v.x, v.y, v.z, v.w);
  }
  *(unsigned int*)(wq + (size_t)row * IN_PAD + k4) = pk;
}

__global__ __launch_bounds__(256)
void gemm_fp8_kernel(const unsigned char* __restrict__ xq,
                     const unsigned char* __restrict__ wq,
                     const float* __restrict__ wscale,
                     const unsigned short* __restrict__ bias,
                     unsigned short* __restrict__ out) {
  __shared__ unsigned char lA[2][TILE * LSTR];
  __shared__ unsigned char lB[2][TILE * LSTR];

  const unsigned int tid  = threadIdx.x;
  const unsigned int lane = tid & 31u;
  const unsigned int wave = tid >> 5;
  const unsigned int half = (lane >> 4) & 1u;
  const unsigned int l16  = lane & 15u;

  const unsigned int n0 = blockIdx.x * TILE;
  const unsigned int m0 = blockIdx.y * TILE;
  const unsigned int waveM = wave >> 1;
  const unsigned int waveN = wave & 1u;

  const unsigned int srow0 = tid >> 3;
  const unsigned int scol  = (tid & 7u) << 4;

  auto stage = [&](unsigned int buf, unsigned int k0) {
#pragma unroll
    for (int pass = 0; pass < 4; ++pass) {
      const unsigned int row = srow0 + 32u * (unsigned int)pass;
      const unsigned char* ga = xq + (size_t)(m0 + row) * IN_PAD + k0 + scol;
      const unsigned char* gb = wq + (size_t)(n0 + row) * IN_PAD + k0 + scol;
      unsigned char* la = &lA[buf][row * LSTR + scol];
      unsigned char* lb = &lB[buf][row * LSTR + scol];
#if USE_ASYNC_COPY
      __builtin_amdgcn_global_load_async_to_lds_b128((g_v4i*)ga, (l_v4i*)la, 0, 0);
      __builtin_amdgcn_global_load_async_to_lds_b128((g_v4i*)gb, (l_v4i*)lb, 0, 0);
#else
      *(v4i_t*)la = *(const v4i_t*)ga;
      *(v4i_t*)lb = *(const v4i_t*)gb;
#endif
    }
  };

  v8f_t acc[8];
#pragma unroll
  for (int i = 0; i < 8; ++i) acc[i] = (v8f_t)0.0f;

  union Frag { v16i_t v; int d[16]; unsigned long long q[8]; };

  stage(0u, 0u);

  const float* wsrow = wscale + (size_t)blockIdx.x * SBLK;
  v2f_t scNext = *(const v2f_t*)wsrow;

  const unsigned int nGroups = (IN_PAD / TILE) / 2;

#pragma unroll 1
  for (unsigned int kb = 0; kb < nGroups; ++kb) {
    const v2f_t sc2 = scNext;
    if (kb + 1 < nGroups) scNext = *(const v2f_t*)(wsrow + (kb + 1) * 2);
    const unsigned int kbase = kb * (2u * TILE);

#pragma unroll
    for (unsigned int j = 0; j < 2; ++j) {
      const unsigned int cur = j & 1u;
#if USE_ASYNC_COPY
      __builtin_amdgcn_s_wait_asynccnt(0);
#endif
      __syncthreads();
      if (j == 0u) {
        stage(cur ^ 1u, kbase + TILE);
      } else if (kb + 1 < nGroups) {
        stage(cur ^ 1u, kbase + 2u * TILE);
      }

      const float sc = sc2[j];

      Frag afr[2];
#pragma unroll
      for (int mt = 0; mt < 2; ++mt) {
        const unsigned char* base = &lA[cur][(waveM * 32u + (unsigned int)mt * 16u + l16) * LSTR];
#pragma unroll
        for (int p = 0; p < 8; ++p) {
          unsigned int kofs = (((unsigned int)p & 1u) << 3) + (((unsigned int)p >> 1) << 5) + (half << 4);
          afr[mt].q[p] = *(const unsigned long long*)(base + kofs);
        }
      }
      Frag bfr[4];
#pragma unroll
      for (int nt = 0; nt < 4; ++nt) {
        const unsigned char* base = &lB[cur][(waveN * 64u + (unsigned int)nt * 16u + l16) * LSTR];
#pragma unroll
        for (int b = 0; b < 4; ++b) {
          unsigned int kofs = ((unsigned int)b << 5) + (half << 4);
          *(v4i_t*)&bfr[nt].d[4 * b] = *(const v4i_t*)(base + kofs);
        }
      }

      const v8f_t zero = (v8f_t)0.0f;
#pragma unroll
      for (int mt = 0; mt < 2; ++mt) {
#pragma unroll
        for (int nt = 0; nt < 4; ++nt) {
          v8f_t part = __builtin_amdgcn_wmma_f32_16x16x128_fp8_fp8(
              afr[mt].v, bfr[nt].v, (short)0, zero, false, false);
          acc[mt * 4 + nt] += part * sc;
        }
      }
    }
  }

#pragma unroll
  for (int mt = 0; mt < 2; ++mt) {
#pragma unroll
    for (int nt = 0; nt < 4; ++nt) {
      unsigned int n = n0 + waveN * 64u + (unsigned int)nt * 16u + l16;
      if (n >= OUT_F) continue;
      const float bv = bf16_to_f32(bias[n]);
      v8f_t a = acc[mt * 4 + nt];
#pragma unroll
      for (int r = 0; r < 8; ++r) {
        unsigned int m = m0 + waveM * 32u + (unsigned int)mt * 16u + half * 8u + (unsigned int)r;
        float v = bf16_to_f32(f32_to_bf16_rne(a[r])) + bv;
        __builtin_nontemporal_store(f32_to_bf16_rne(v), out + (size_t)m * OUT_F + n);
      }
    }
  }
}

extern "C" void kernel_launch(void* const* d_in, const int* in_sizes, int n_in,
                              void* d_out, int out_size, void* d_ws, size_t ws_size,
                              hipStream_t stream) {
  const float*          x      = (const float*)d_in[0];
  const float*          w      = (const float*)d_in[1];
  const float*          wscale = (const float*)d_in[2];
  const unsigned short* bias   = (const unsigned short*)d_in[3];
  unsigned short*       out    = (unsigned short*)d_out;

  unsigned char* xq = (unsigned char*)d_ws;
  unsigned char* wq = xq + (size_t)BATCH * IN_PAD;

  quant_x_kernel<<<(BATCH * (IN_PAD / 4)) / 256, 256, 0, stream>>>(x, xq);
  quant_w_kernel<<<(N_PAD * (IN_PAD / 4)) / 256, 256, 0, stream>>>(w, wq);
  dim3 grid(N_PAD / TILE, BATCH / TILE);
  gemm_fp8_kernel<<<grid, 256, 0, stream>>>(xq, wq, wscale, bias, out);
}
